// SparseMoE_2611340116275
// MI455X (gfx1250) — compile-verified, matrix operand order checked by reading, not run on hardware
//
#include <hip/hip_runtime.h>
#include <math.h>

typedef __attribute__((ext_vector_type(16))) _Float16 h16x16;
typedef __attribute__((ext_vector_type(8)))  _Float16 h16x8;
typedef __attribute__((ext_vector_type(4)))  _Float16 h16x4;
typedef __attribute__((ext_vector_type(8)))  float    f32x8;

constexpr int D_  = 2048;   // model dim
constexpr int F_  = 512;    // ffn dim
constexpr int E_  = 8;      // routed experts
constexpr int SH_ = 2;      // shared experts
constexpr int N_  = 8192;   // tokens = B*S
constexpr int K_  = 2;      // top-k

constexpr int TMB = 128;    // block tile M
constexpr int TNB = 64;     // block tile N
constexpr int TK  = 32;     // k-step (f16 WMMA K)
constexpr int PA  = 40;     // LDS tile pitch in halves (80B: 16B-aligned, conflict-spread)
constexpr int LAS = TMB * PA;   // halves per A LDS buffer
constexpr int LBS = TNB * PA;   // halves per B LDS buffer

// ---------------------------------------------------------------- helpers

__device__ __forceinline__ float gelu_f(float v) {
    return 0.5f * v * (1.0f + erff(v * 0.70710678118654752440f));
}

union H16U { h16x16 v; h16x8 h[2]; };

// Build A/B fragment for v_wmma_f32_16x16x32_f16.
// 16-bit A layout (ISA 7.12.2): lanes 0-15 hold K {0..7, 16..23}, lanes 16-31 hold K {8..15, 24..31}.
// B tiles are stored n-major in LDS, so the same pattern applies with row = n.
__device__ __forceinline__ h16x16 ld_frag(const _Float16* rowbase, int hsel) {
    H16U u;
    u.h[0] = *(const h16x8*)(rowbase + hsel * 8);        // K = hsel*8 .. +7
    u.h[1] = *(const h16x8*)(rowbase + 16 + hsel * 8);   // K = 16 + hsel*8 ..
    return u.v;
}

#define WMMA_F16(A, B, C) \
    __builtin_amdgcn_wmma_f32_16x16x32_f16(false, (A), false, (B), (short)0, (C), false, false)

// Async global->LDS 16B copy (gfx1250, tracked with ASYNCcnt).
// VDST VGPR holds the wave-relative LDS byte address (= low 32 bits of the
// flat pointer to a __shared__ object); VADDR is the 64-bit global address.
__device__ __forceinline__ void async_ld16(unsigned lds_off, const void* gaddr) {
    asm volatile("global_load_async_to_lds_b128 %0, %1, off"
                 :: "v"(lds_off), "v"(gaddr) : "memory");
}
#define WAIT_ASYNCCNT(n) asm volatile("s_wait_asynccnt " #n ::: "memory")

// Triple-buffered, distance-2 async pipeline. 128x64 block tile, 8 waves in
// 4(M) x 2(N), each wave 32x32 (4 accumulators, 2 A-frags x 2 B-frags, 4 WMMA/step).
// Iteration s DMAs buffer s+2 and waits asynccnt<=3 before its barrier, which
// (in-order per-wave async completion) guarantees the batch for buffer s+1 has
// landed while leaving this iteration's 3 DMAs in flight — a full iteration of
// WMMA work covers each DMA. Tail iterations (nothing issued) drain to 0.
// Overwrite safety: the batch issued at s targets the buffer consumed at s-1;
// every wave's frag ds_loads complete before its WMMA, hence before barrier(s-1).
__device__ __forceinline__ void gemm_mainloop(
    const _Float16* aRow0, const _Float16* aRow1, const _Float16* bRow, int Ktot,
    _Float16* lA, _Float16* lB, _Float16* lar0, _Float16* lar1, _Float16* lbr,
    int wm, int wn, int r16, int hsel,
    f32x8& acc00, f32x8& acc01, f32x8& acc10, f32x8& acc11)
{
    const unsigned oA0 = (unsigned)(uintptr_t)lar0;
    const unsigned oA1 = (unsigned)(uintptr_t)lar1;
    const unsigned oB  = (unsigned)(uintptr_t)lbr;
    const int nsteps = Ktot / TK;   // >= 16 at every call site
    // prologue: prefetch buffers 0 and 1
    async_ld16(oA0, aRow0);
    async_ld16(oA1, aRow1);
    async_ld16(oB,  bRow);
    async_ld16(oA0 + (LAS * 2), aRow0 + TK);
    async_ld16(oA1 + (LAS * 2), aRow1 + TK);
    async_ld16(oB  + (LBS * 2), bRow  + TK);
    WAIT_ASYNCCNT(0x3);            // buffer 0 complete; buffer 1 may be in flight
    __syncthreads();
    for (int s = 0; s < nsteps; s++) {
        const int cur = s % 3;
        const bool pfed = (s + 2 < nsteps);
        if (pfed) {
            const int pf = (s + 2) % 3;
            const int kk = (s + 2) * TK;
            async_ld16(oA0 + pf * (LAS * 2), aRow0 + kk);
            async_ld16(oA1 + pf * (LAS * 2), aRow1 + kk);
            async_ld16(oB  + pf * (LBS * 2), bRow  + kk);
        }
        const _Float16* lAc = lA + cur * LAS;
        const _Float16* lBc = lB + cur * LBS;
        h16x16 af0 = ld_frag(lAc + (wm + r16) * PA, hsel);
        h16x16 af1 = ld_frag(lAc + (wm + 16 + r16) * PA, hsel);
        h16x16 bf0 = ld_frag(lBc + (wn + r16) * PA, hsel);
        h16x16 bf1 = ld_frag(lBc + (wn + 16 + r16) * PA, hsel);
        acc00 = WMMA_F16(af0, bf0, acc00);
        acc01 = WMMA_F16(af0, bf1, acc01);
        acc10 = WMMA_F16(af1, bf0, acc10);
        acc11 = WMMA_F16(af1, bf1, acc11);
        if (pfed) { WAIT_ASYNCCNT(0x3); }   // batch for buffer s+1 has landed
        else      { WAIT_ASYNCCNT(0x0); }   // tail: only that batch outstanding
        __syncthreads();
    }
}

// ---------------------------------------------------------------- converts

__global__ __launch_bounds__(256) void cvt_x_kernel(const float* __restrict__ x,
                                                    _Float16* __restrict__ xh) {
    size_t i = ((size_t)blockIdx.x * 256 + threadIdx.x) * 4;
    float4 v = *(const float4*)(x + i);
    h16x4 o = { (_Float16)v.x, (_Float16)v.y, (_Float16)v.z, (_Float16)v.w };
    *(h16x4*)(xh + i) = o;
}

// in[z][r][c] (f32, RxC) -> out[z*obs + c*OS + r] (f16)
__global__ __launch_bounds__(256) void transpose_cvt_kernel(
    const float* __restrict__ in, _Float16* __restrict__ out,
    int R, int C, long long ibs, long long obs, int OS)
{
    __shared__ float t[32][33];
    int c0 = blockIdx.x * 32, r0 = blockIdx.y * 32, z = blockIdx.z;
    const float* ip = in + (long long)z * ibs;
#pragma unroll
    for (int j = 0; j < 4; j++)
        t[threadIdx.y + 8 * j][threadIdx.x] =
            ip[(long long)(r0 + threadIdx.y + 8 * j) * C + c0 + threadIdx.x];
    __syncthreads();
    _Float16* op = out + (long long)z * obs;
#pragma unroll
    for (int j = 0; j < 4; j++)
        op[(long long)(c0 + threadIdx.y + 8 * j) * OS + r0 + threadIdx.x] =
            (_Float16)t[threadIdx.x][threadIdx.y + 8 * j];
}

// ---------------------------------------------------------------- router

__global__ __launch_bounds__(256) void router_kernel(
    const float* __restrict__ x, const float* __restrict__ rw, const float* __restrict__ rb,
    int2* __restrict__ ti, float2* __restrict__ tg,
    int* __restrict__ counts, int* __restrict__ cslot)
{
    __shared__ float lrw[E_ * 512];
    int tid = threadIdx.x, lane = tid & 31, wid = tid >> 5;
    int tbase = blockIdx.x * 32;          // 32 tokens per block, 4 per wave
    float acc[4][E_];
#pragma unroll
    for (int tk = 0; tk < 4; tk++)
#pragma unroll
        for (int e = 0; e < E_; e++) acc[tk][e] = 0.f;

    for (int c0 = 0; c0 < D_; c0 += 512) {
        for (int i = tid; i < E_ * 512; i += 256)
            lrw[i] = rw[(i >> 9) * D_ + c0 + (i & 511)];
        __syncthreads();
#pragma unroll
        for (int tk = 0; tk < 4; tk++) {
            int t = tbase + wid * 4 + tk;
            const float* xp = x + (size_t)t * D_ + c0;
            for (int d = lane; d < 512; d += 32) {
                float xv = xp[d];
#pragma unroll
                for (int e = 0; e < E_; e++) acc[tk][e] += xv * lrw[e * 512 + d];
            }
        }
        __syncthreads();
    }

#pragma unroll
    for (int tk = 0; tk < 4; tk++) {
        int t = tbase + wid * 4 + tk;
        float v[E_];
#pragma unroll
        for (int e = 0; e < E_; e++) {
            float s = acc[tk][e];
            for (int off = 16; off > 0; off >>= 1) s += __shfl_xor(s, off, 32);
            v[e] = s + rb[e];
        }
        if (lane == 0) {
            int i0 = 0; float v0 = v[0];
#pragma unroll
            for (int e = 1; e < E_; e++) if (v[e] > v0) { v0 = v[e]; i0 = e; }
            int i1 = -1; float v1 = -3.4e38f;
#pragma unroll
            for (int e = 0; e < E_; e++) if (e != i0 && v[e] > v1) { v1 = v[e]; i1 = e; }
            float ex = expf(v1 - v0);
            float g0 = 1.f / (1.f + ex), g1 = ex * g0;
            ti[t] = make_int2(i0, i1);
            tg[t] = make_float2(g0, g1);
            atomicAdd(&counts[i0], 1);
            atomicAdd(&counts[i1], 1);
            if (i0 == 0) atomicAdd(&cslot[0], 1);
            if (i1 == 1) atomicAdd(&cslot[1], 1);
        }
    }
}

__global__ void offsets_kernel(const int* __restrict__ counts, int* __restrict__ offs) {
    if (threadIdx.x == 0) {
        int s = 0;
        for (int e = 0; e < E_; e++) { offs[e] = s; s += counts[e]; }
    }
}

__global__ __launch_bounds__(256) void scatter_kernel(
    const int2* __restrict__ ti, const float2* __restrict__ tg,
    const int* __restrict__ offs, int* __restrict__ cursors,
    int* __restrict__ toks, float* __restrict__ gts)
{
    int t = blockIdx.x * 256 + threadIdx.x;
    if (t >= N_) return;
    int2 i = ti[t]; float2 g = tg[t];
    int p0 = atomicAdd(&cursors[i.x], 1);
    toks[offs[i.x] + p0] = t; gts[offs[i.x] + p0] = g.x;
    int p1 = atomicAdd(&cursors[i.y], 1);
    toks[offs[i.y] + p1] = t; gts[offs[i.y] + p1] = g.y;
}

// ---------------------------------------------------------------- shared experts

__global__ __launch_bounds__(256) void ffn1_shared_kernel(
    const _Float16* __restrict__ xh, const _Float16* __restrict__ sw1t,
    const float* __restrict__ sb1, _Float16* __restrict__ hbuf)
{
    int sh = blockIdx.z;
    int m0 = blockIdx.x * TMB, n0 = blockIdx.y * TNB;
    __shared__ _Float16 lA[3 * LAS], lB[3 * LBS];
    int tid = threadIdx.x;
    int rA = tid >> 2, lc = tid & 3;
    const _Float16* aRow0 = xh + (size_t)(m0 + rA) * D_ + lc * 8;
    const _Float16* aRow1 = xh + (size_t)(m0 + rA + 64) * D_ + lc * 8;
    const _Float16* bRow  = sw1t + (size_t)sh * F_ * D_ + (size_t)(n0 + rA) * D_ + lc * 8;
    _Float16* lar0 = lA + rA * PA + lc * 8;
    _Float16* lar1 = lA + (rA + 64) * PA + lc * 8;
    _Float16* lbr  = lB + rA * PA + lc * 8;
    int lane = tid & 31, wid = tid >> 5;
    int wm = (wid & 3) * 32, wn = (wid >> 2) * 32;
    int r16 = lane & 15, hsel = lane >> 4;
    f32x8 acc00 = {0,0,0,0,0,0,0,0}, acc01 = acc00, acc10 = acc00, acc11 = acc00;
    gemm_mainloop(aRow0, aRow1, bRow, D_, lA, lB, lar0, lar1, lbr,
                  wm, wn, r16, hsel, acc00, acc01, acc10, acc11);

    int col = n0 + wn + r16;
    int rb0 = m0 + wm + hsel * 8;
    float b0 = sb1[sh * F_ + col], b1 = sb1[sh * F_ + col + 16];
#pragma unroll
    for (int i = 0; i < 8; i++) {
        size_t base0 = (size_t)(rb0 + i) * (2 * F_) + sh * F_;
        size_t base1 = (size_t)(rb0 + 16 + i) * (2 * F_) + sh * F_;
        hbuf[base0 + col]      = (_Float16)gelu_f(acc00[i] + b0);
        hbuf[base0 + col + 16] = (_Float16)gelu_f(acc01[i] + b1);
        hbuf[base1 + col]      = (_Float16)gelu_f(acc10[i] + b0);
        hbuf[base1 + col + 16] = (_Float16)gelu_f(acc11[i] + b1);
    }
}

__global__ __launch_bounds__(256) void ffn2_shared_kernel(
    const _Float16* __restrict__ hbuf, const _Float16* __restrict__ sw2t,
    const float* __restrict__ sb2, float* __restrict__ out)
{
    int m0 = blockIdx.x * TMB, n0 = blockIdx.y * TNB;
    __shared__ _Float16 lA[3 * LAS], lB[3 * LBS];
    int tid = threadIdx.x;
    int rA = tid >> 2, lc = tid & 3;
    const _Float16* aRow0 = hbuf + (size_t)(m0 + rA) * (2 * F_) + lc * 8;
    const _Float16* aRow1 = hbuf + (size_t)(m0 + rA + 64) * (2 * F_) + lc * 8;
    const _Float16* bRow  = sw2t + (size_t)(n0 + rA) * (2 * F_) + lc * 8;
    _Float16* lar0 = lA + rA * PA + lc * 8;
    _Float16* lar1 = lA + (rA + 64) * PA + lc * 8;
    _Float16* lbr  = lB + rA * PA + lc * 8;
    int lane = tid & 31, wid = tid >> 5;
    int wm = (wid & 3) * 32, wn = (wid >> 2) * 32;
    int r16 = lane & 15, hsel = lane >> 4;
    f32x8 acc00 = {0,0,0,0,0,0,0,0}, acc01 = acc00, acc10 = acc00, acc11 = acc00;
    gemm_mainloop(aRow0, aRow1, bRow, 2 * F_, lA, lB, lar0, lar1, lbr,
                  wm, wn, r16, hsel, acc00, acc01, acc10, acc11);

    int col = n0 + wn + r16;
    int rb0 = m0 + wm + hsel * 8;
    float b0 = sb2[col] + sb2[D_ + col];
    float b1 = sb2[col + 16] + sb2[D_ + col + 16];
#pragma unroll
    for (int i = 0; i < 8; i++) {
        size_t base0 = (size_t)(rb0 + i) * D_;
        size_t base1 = (size_t)(rb0 + 16 + i) * D_;
        out[base0 + col]      = acc00[i] + b0;
        out[base0 + col + 16] = acc01[i] + b1;
        out[base1 + col]      = acc10[i] + b0;
        out[base1 + col + 16] = acc11[i] + b1;
    }
}

// ---------------------------------------------------------------- routed experts

__global__ __launch_bounds__(256) void ffn1_routed_kernel(
    const _Float16* __restrict__ xh, const _Float16* __restrict__ w1t,
    const float* __restrict__ eb1, const int* __restrict__ counts,
    const int* __restrict__ offs, const int* __restrict__ toks,
    _Float16* __restrict__ hbuf)
{
    int e = blockIdx.z;
    int cnt = counts[e];
    int m0 = blockIdx.x * TMB;
    if (m0 >= cnt) return;
    int off = offs[e];
    int n0 = blockIdx.y * TNB;
    __shared__ _Float16 lA[3 * LAS], lB[3 * LBS];
    __shared__ int s_tok[TMB];
    int tid = threadIdx.x;
    if (tid < TMB) {
        int r = m0 + tid; if (r >= cnt) r = cnt - 1;
        s_tok[tid] = toks[off + r];
    }
    __syncthreads();
    int rA = tid >> 2, lc = tid & 3;
    const _Float16* aRow0 = xh + (size_t)s_tok[rA] * D_ + lc * 8;
    const _Float16* aRow1 = xh + (size_t)s_tok[rA + 64] * D_ + lc * 8;
    const _Float16* bRow  = w1t + (size_t)e * F_ * D_ + (size_t)(n0 + rA) * D_ + lc * 8;
    _Float16* lar0 = lA + rA * PA + lc * 8;
    _Float16* lar1 = lA + (rA + 64) * PA + lc * 8;
    _Float16* lbr  = lB + rA * PA + lc * 8;
    int lane = tid & 31, wid = tid >> 5;
    int wm = (wid & 3) * 32, wn = (wid >> 2) * 32;
    int r16 = lane & 15, hsel = lane >> 4;
    f32x8 acc00 = {0,0,0,0,0,0,0,0}, acc01 = acc00, acc10 = acc00, acc11 = acc00;
    gemm_mainloop(aRow0, aRow1, bRow, D_, lA, lB, lar0, lar1, lbr,
                  wm, wn, r16, hsel, acc00, acc01, acc10, acc11);

    int col = n0 + wn + r16;
    const float* b = eb1 + e * F_;
    float b0 = b[col], b1 = b[col + 16];
#pragma unroll
    for (int i = 0; i < 8; i++) {
        int l0 = wm + hsel * 8 + i;
        int l1 = l0 + 16;
        if (m0 + l0 < cnt) {
            size_t base = (size_t)(off + m0 + l0) * F_;
            hbuf[base + col]      = (_Float16)gelu_f(acc00[i] + b0);
            hbuf[base + col + 16] = (_Float16)gelu_f(acc01[i] + b1);
        }
        if (m0 + l1 < cnt) {
            size_t base = (size_t)(off + m0 + l1) * F_;
            hbuf[base + col]      = (_Float16)gelu_f(acc10[i] + b0);
            hbuf[base + col + 16] = (_Float16)gelu_f(acc11[i] + b1);
        }
    }
}

__global__ __launch_bounds__(256) void ffn2_routed_kernel(
    const _Float16* __restrict__ hbuf, const _Float16* __restrict__ w2t,
    const float* __restrict__ eb2, const int* __restrict__ counts,
    const int* __restrict__ offs, const int* __restrict__ toks,
    const float* __restrict__ gts, float* __restrict__ out)
{
    int e = blockIdx.z;
    int cnt = counts[e];
    int m0 = blockIdx.x * TMB;
    if (m0 >= cnt) return;
    int off = offs[e];
    int n0 = blockIdx.y * TNB;
    __shared__ _Float16 lA[3 * LAS], lB[3 * LBS];
    __shared__ int s_tok[TMB];
    __shared__ float s_gate[TMB];
    int tid = threadIdx.x;
    if (tid < TMB) {
        int r = m0 + tid; if (r >= cnt) r = cnt - 1;
        s_tok[tid]  = toks[off + r];
        s_gate[tid] = gts[off + r];
    }
    __syncthreads();
    int rA = tid >> 2, lc = tid & 3;
    int ar0 = m0 + rA;      if (ar0 >= cnt) ar0 = cnt - 1;
    int ar1 = m0 + rA + 64; if (ar1 >= cnt) ar1 = cnt - 1;
    const _Float16* aRow0 = hbuf + (size_t)(off + ar0) * F_ + lc * 8;
    const _Float16* aRow1 = hbuf + (size_t)(off + ar1) * F_ + lc * 8;
    const _Float16* bRow  = w2t + (size_t)e * D_ * F_ + (size_t)(n0 + rA) * F_ + lc * 8;
    _Float16* lar0 = lA + rA * PA + lc * 8;
    _Float16* lar1 = lA + (rA + 64) * PA + lc * 8;
    _Float16* lbr  = lB + rA * PA + lc * 8;
    int lane = tid & 31, wid = tid >> 5;
    int wm = (wid & 3) * 32, wn = (wid >> 2) * 32;
    int r16 = lane & 15, hsel = lane >> 4;
    f32x8 acc00 = {0,0,0,0,0,0,0,0}, acc01 = acc00, acc10 = acc00, acc11 = acc00;
    gemm_mainloop(aRow0, aRow1, bRow, F_, lA, lB, lar0, lar1, lbr,
                  wm, wn, r16, hsel, acc00, acc01, acc10, acc11);

    int col = n0 + wn + r16;
    float b0 = eb2[e * D_ + col], b1 = eb2[e * D_ + col + 16];
#pragma unroll
    for (int i = 0; i < 8; i++) {
        int l0 = wm + hsel * 8 + i;
        int l1 = l0 + 16;
        if (m0 + l0 < cnt) {
            float g = s_gate[l0];
            size_t base = (size_t)s_tok[l0] * D_;
            atomicAdd(&out[base + col],      g * (acc00[i] + b0));
            atomicAdd(&out[base + col + 16], g * (acc01[i] + b1));
        }
        if (m0 + l1 < cnt) {
            float g = s_gate[l1];
            size_t base = (size_t)s_tok[l1] * D_;
            atomicAdd(&out[base + col],      g * (acc10[i] + b0));
            atomicAdd(&out[base + col + 16], g * (acc11[i] + b1));
        }
    }
}

// ---------------------------------------------------------------- aux loss

__global__ void loss_kernel(const int* __restrict__ cslot, float* __restrict__ out_loss) {
    if (threadIdx.x == 0) {
        const float nk = (float)(N_ * K_);
        float l = 0.f;
        for (int e = 0; e < E_; e++) {
            float c = (e == 0) ? (float)cslot[0] : (e == 1) ? (float)cslot[1] : 0.f;
            float p = c / nk + 1e-8f;
            l -= p * logf(p);
        }
        *out_loss = l;
    }
}

// ---------------------------------------------------------------- launch

extern "C" void kernel_launch(void* const* d_in, const int* in_sizes, int n_in,
                              void* d_out, int out_size, void* d_ws, size_t ws_size,
                              hipStream_t stream) {
    const float* x    = (const float*)d_in[0];
    const float* rw   = (const float*)d_in[1];
    const float* rb   = (const float*)d_in[2];
    const float* ew1  = (const float*)d_in[3];
    const float* eb1  = (const float*)d_in[4];
    const float* ew2  = (const float*)d_in[5];
    const float* eb2  = (const float*)d_in[6];
    const float* sw1  = (const float*)d_in[7];
    const float* sb1  = (const float*)d_in[8];
    const float* sw2  = (const float*)d_in[9];
    const float* sb2  = (const float*)d_in[10];
    float* out = (float*)d_out;

    // workspace carve (256B aligned)
    char* p = (char*)d_ws;
    auto carve = [&](size_t bytes) { char* r = p; p += (bytes + 255) & ~(size_t)255; return (void*)r; };
    _Float16* xh    = (_Float16*)carve((size_t)N_ * D_ * 2);
    _Float16* w1t   = (_Float16*)carve((size_t)E_ * D_ * F_ * 2);   // (E, F, D)
    _Float16* w2t   = (_Float16*)carve((size_t)E_ * F_ * D_ * 2);   // (E, D, F)
    _Float16* sw1t  = (_Float16*)carve((size_t)SH_ * D_ * F_ * 2);  // (SH, F, D)
    _Float16* sw2t  = (_Float16*)carve((size_t)D_ * 2 * F_ * 2);    // (D, 2F)
    _Float16* hbuf  = (_Float16*)carve((size_t)2 * N_ * F_ * 2);    // shared: (N,2F); routed: (2N,F)
    int*      toks  = (int*)carve((size_t)2 * N_ * 4);
    float*    gts   = (float*)carve((size_t)2 * N_ * 4);
    int2*     ti    = (int2*)carve((size_t)N_ * 8);
    float2*   tg    = (float2*)carve((size_t)N_ * 8);
    int*      meta  = (int*)carve(32 * 4);
    int* counts  = meta;        // [8]
    int* offs    = meta + 8;    // [8]
    int* cursors = meta + 16;   // [8]
    int* cslot   = meta + 24;   // [2]

    hipMemsetAsync(meta, 0, 32 * 4, stream);

    // f16 conversions / transposes
    cvt_x_kernel<<<(N_ * D_) / 1024, 256, 0, stream>>>(x, xh);
    transpose_cvt_kernel<<<dim3(F_ / 32, D_ / 32, E_), dim3(32, 8), 0, stream>>>(
        ew1, w1t, D_, F_, (long long)D_ * F_, (long long)F_ * D_, D_);
    transpose_cvt_kernel<<<dim3(D_ / 32, F_ / 32, E_), dim3(32, 8), 0, stream>>>(
        ew2, w2t, F_, D_, (long long)F_ * D_, (long long)D_ * F_, F_);
    transpose_cvt_kernel<<<dim3(F_ / 32, D_ / 32, SH_), dim3(32, 8), 0, stream>>>(
        sw1, sw1t, D_, F_, (long long)D_ * F_, (long long)F_ * D_, D_);
    transpose_cvt_kernel<<<dim3(D_ / 32, F_ / 32, SH_), dim3(32, 8), 0, stream>>>(
        sw2, sw2t, F_, D_, (long long)F_ * D_, (long long)F_, 2 * F_);

    // shared experts (dense), initializes out
    ffn1_shared_kernel<<<dim3(N_ / TMB, F_ / TNB, SH_), 256, 0, stream>>>(xh, sw1t, sb1, hbuf);
    ffn2_shared_kernel<<<dim3(N_ / TMB, D_ / TNB, 1), 256, 0, stream>>>(hbuf, sw2t, sb2, out);

    // routing
    router_kernel<<<N_ / 32, 256, 0, stream>>>(x, rw, rb, ti, tg, counts, cslot);
    offsets_kernel<<<1, 32, 0, stream>>>(counts, offs);
    scatter_kernel<<<N_ / 256, 256, 0, stream>>>(ti, tg, offs, cursors, toks, gts);

    // routed experts (grouped GEMMs with early-exit tiles), reuses hbuf
    ffn1_routed_kernel<<<dim3(N_ / TMB, F_ / TNB, E_), 256, 0, stream>>>(
        xh, w1t, eb1, counts, offs, toks, hbuf);
    ffn2_routed_kernel<<<dim3(N_ / TMB, D_ / TNB, E_), 256, 0, stream>>>(
        hbuf, w2t, eb2, counts, offs, toks, gts, out);

    // aux loss scalar
    loss_kernel<<<1, 32, 0, stream>>>(cslot, out + (size_t)N_ * D_);
}
